// APPNPNet_80075370266743
// MI455X (gfx1250) — matrix operand order repaired and checked by reading, not compiled, not run on hardware
//
#include <hip/hip_runtime.h>
#include <hip/hip_bf16.h>
#include <stdint.h>

typedef _Float16 f16;
typedef __attribute__((ext_vector_type(16))) _Float16 v16h;
typedef __attribute__((ext_vector_type(8)))  _Float16 v8h;
typedef __attribute__((ext_vector_type(8)))  float    v8f;

#define IN_CH  512
#define HID_CH 256
#define OUT_CH 32
#define K_ITER 10
#define ALPHA  0.1f

#define M_TILES 5              // 16-row tiles per block (one B-frag feeds 5 WMMAs)
#define ROWS    (16 * M_TILES) // 80 rows per block; 100000 = 1250 * 80

// ---------------------------------------------------------------------------
// K0: transpose + f32->f16 convert weights: W1 [512][256] -> W1T [256][512],
//     W2 [256][32] -> W2T [32][256]
// ---------------------------------------------------------------------------
__global__ void transpose_weights(const float* __restrict__ W1,
                                  const float* __restrict__ W2,
                                  f16* __restrict__ W1T, f16* __restrict__ W2T) {
    int i = blockIdx.x * blockDim.x + threadIdx.x;
    if (i < IN_CH * HID_CH) {
        int k = i / HID_CH, n = i % HID_CH;
        W1T[n * IN_CH + k] = (f16)W1[i];
    }
    if (i < HID_CH * OUT_CH) {
        int k = i / OUT_CH, n = i % OUT_CH;
        W2T[n * HID_CH + k] = (f16)W2[i];
    }
}

// ---------------------------------------------------------------------------
// Degree / norm precompute (self-loop included as +1)
// ---------------------------------------------------------------------------
__global__ void init_deg(float* __restrict__ deg, int* __restrict__ cnt, int N) {
    int i = blockIdx.x * blockDim.x + threadIdx.x;
    if (i < N) { deg[i] = 1.0f; cnt[i] = 0; }
}

__global__ void count_edges(const int* __restrict__ colE, float* __restrict__ deg,
                            int* __restrict__ cnt, int E) {
    int e = blockIdx.x * blockDim.x + threadIdx.x;
    if (e < E) {
        int c = colE[e];
        atomicAdd(&deg[c], 1.0f);
        atomicAdd(&cnt[c], 1);
    }
}

__global__ void compute_dinv(const float* __restrict__ deg, float* __restrict__ dinv, int N) {
    int i = blockIdx.x * blockDim.x + threadIdx.x;
    if (i < N) dinv[i] = rsqrtf(deg[i]);   // deg >= 1 always (self loop)
}

// ---------------------------------------------------------------------------
// Exclusive prefix sum of cnt -> rowptr (block scan -> scan of block sums -> add)
// ---------------------------------------------------------------------------
__global__ void scan_blocks(const int* __restrict__ cnt, int* __restrict__ rowptr,
                            int* __restrict__ bsum, int N) {
    __shared__ int s[256];
    int t = threadIdx.x;
    int i = blockIdx.x * 256 + t;
    int v = (i < N) ? cnt[i] : 0;
    s[t] = v;
    __syncthreads();
    for (int off = 1; off < 256; off <<= 1) {
        int u = (t >= off) ? s[t - off] : 0;
        __syncthreads();
        s[t] += u;
        __syncthreads();
    }
    if (i < N) rowptr[i] = s[t] - v;           // exclusive, block-local
    if (t == 255) bsum[blockIdx.x] = s[255];   // block total
}

__global__ void scan_sums(const int* __restrict__ bsum, int* __restrict__ boff, int nb) {
    __shared__ int s[512];
    int t = threadIdx.x;
    int v = (t < nb) ? bsum[t] : 0;
    s[t] = v;
    __syncthreads();
    for (int off = 1; off < 512; off <<= 1) {
        int u = (t >= off) ? s[t - off] : 0;
        __syncthreads();
        s[t] += u;
        __syncthreads();
    }
    if (t < nb) boff[t] = s[t] - v;            // exclusive over blocks
}

__global__ void add_offsets(int* __restrict__ rowptr, const int* __restrict__ boff,
                            int* __restrict__ cursor, int N, int E) {
    int i = blockIdx.x * blockDim.x + threadIdx.x;
    if (i < N) {
        int r = rowptr[i] + boff[i >> 8];
        rowptr[i] = r;
        cursor[i] = r;
    }
    if (i == 0) rowptr[N] = E;
}

// ---------------------------------------------------------------------------
// CSR fill: bucket edges by destination (col); store (src, norm) packed
// ---------------------------------------------------------------------------
__global__ void fill_csr(const int* __restrict__ rowE, const int* __restrict__ colE,
                         const float* __restrict__ dinv, int* __restrict__ cursor,
                         int2* __restrict__ edges, int E) {
    int e = blockIdx.x * blockDim.x + threadIdx.x;
    if (e < E) {
        int sidx = rowE[e];
        int d = colE[e];
        int pos = atomicAdd(&cursor[d], 1);
        edges[pos] = make_int2(sidx, __float_as_int(dinv[sidx] * dinv[d]));
    }
}

// ---------------------------------------------------------------------------
// K1: fused MLP via WMMA. Block = 512 threads = 16 waves, 80 rows (5 M-tiles).
//   Phase 1: wave w computes hidden cols [16w,16w+16) for all 5 M-tiles;
//            one B-fragment of W1T feeds 5 WMMAs (5x L2 traffic reduction).
//   Phase 2: waves 0..9 each compute one (M-tile, 16-col chunk) of the output.
// Writes H0 and HA (propagation initial state).
// ---------------------------------------------------------------------------
#define AX_STRIDE 520   // 512 + pad: rotates LDS banks per row
#define HT_STRIDE 264   // 256 + pad

__global__ __launch_bounds__(512)
void mlp_wmma(const float* __restrict__ x, const f16* __restrict__ W1T,
              const f16* __restrict__ W2T, const float* __restrict__ b1,
              const float* __restrict__ b2, float* __restrict__ H0,
              float* __restrict__ HA, int N) {
    __shared__ f16 Ax[ROWS * AX_STRIDE];   // x tile, f16   (~83 KB)
    __shared__ f16 Ht[ROWS * HT_STRIDE];   // hidden tile   (~42 KB)

    const int tid  = threadIdx.x;
    const int wv   = tid >> 5;
    const int lane = tid & 31;
    const int m0   = blockIdx.x * ROWS;

    // stage ROWSx512 f32 x-tile -> f16 LDS (coalesced float4 loads, row clamp)
    for (int i = tid; i < (ROWS * IN_CH) / 4; i += 512) {
        int r  = i >> 7;        // IN_CH/4 = 128 float4 per row
        int cq = i & 127;
        int gr = m0 + r;
        if (gr >= N) gr = N - 1;
        float4 v = *((const float4*)(x + (size_t)gr * IN_CH) + cq);
        f16* p = &Ax[r * AX_STRIDE + cq * 4];
        p[0] = (f16)v.x; p[1] = (f16)v.y; p[2] = (f16)v.z; p[3] = (f16)v.w;
    }
    __syncthreads();

    const int lhalf = lane >> 4;   // 0: lanes 0-15, 1: lanes 16-31
    const int l16   = lane & 15;

    // ---- Phase 1: hidden = relu(x @ W1 + b1) ----
    {
        v8f c[M_TILES];
#pragma unroll
        for (int t = 0; t < M_TILES; ++t)
            c[t] = (v8f){0.f, 0.f, 0.f, 0.f, 0.f, 0.f, 0.f, 0.f};

        const int n = wv * 16 + l16;          // hidden column this lane covers
        const f16* bcol = W1T + (size_t)n * IN_CH;
#pragma unroll 2
        for (int kk = 0; kk < IN_CH; kk += 32) {
            // B fragment: 16 contiguous K of column n
            // (lanes 0-15: K=kk.., lanes 16-31: K=kk+16..)
            v16h b = *(const v16h*)(bcol + kk + lhalf * 16);
#pragma unroll
            for (int t = 0; t < M_TILES; ++t) {
                // A fragment: row = t*16+l16,
                // K = kk + {0..15} (lo half) / {16..31} (hi half)
                const f16* ar = &Ax[(t * 16 + l16) * AX_STRIDE + kk + lhalf * 16];
                v8h alo = *(const v8h*)ar;
                v8h ahi = *(const v8h*)(ar + 8);
                v16h a = __builtin_shufflevector(alo, ahi,
                          0,1,2,3,4,5,6,7,8,9,10,11,12,13,14,15);
                c[t] = __builtin_amdgcn_wmma_f32_16x16x32_f16(false, a, false, b,
                                                              (short)0, c[t],
                                                              false, false);
            }
        }
        float bias = b1[n];
#pragma unroll
        for (int t = 0; t < M_TILES; ++t) {
#pragma unroll
            for (int r = 0; r < 8; ++r) {
                float h = c[t][r] + bias;
                h = h > 0.f ? h : 0.f;                              // relu
                Ht[(t * 16 + r + lhalf * 8) * HT_STRIDE + n] = (f16)h;
            }
        }
    }
    __syncthreads();

    // ---- Phase 2: out = Ht @ W2 + b2 (5 M-tiles x 2 col-chunks -> waves 0..9) ----
    if (wv < 2 * M_TILES) {
        const int mt = wv >> 1;        // which M-tile
        const int nc = wv & 1;         // which 16-col chunk of OUT_CH
        v8f c = {0.f, 0.f, 0.f, 0.f, 0.f, 0.f, 0.f, 0.f};
        const int n = nc * 16 + l16;
        const f16* bcol = W2T + (size_t)n * HID_CH;
#pragma unroll 2
        for (int kk = 0; kk < HID_CH; kk += 32) {
            const f16* ar = &Ht[(mt * 16 + l16) * HT_STRIDE + kk + lhalf * 16];
            v8h alo = *(const v8h*)ar;
            v8h ahi = *(const v8h*)(ar + 8);
            v16h a = __builtin_shufflevector(alo, ahi,
                      0,1,2,3,4,5,6,7,8,9,10,11,12,13,14,15);
            v16h b = *(const v16h*)(bcol + kk + lhalf * 16);
            c = __builtin_amdgcn_wmma_f32_16x16x32_f16(false, a, false, b, (short)0, c,
                                                       false, false);
        }
        float bias = b2[n];
#pragma unroll
        for (int r = 0; r < 8; ++r) {
            int row = m0 + mt * 16 + r + lhalf * 8;
            if (row < N) {
                float h = c[r] + bias;
                size_t o = (size_t)row * OUT_CH + n;
                H0[o] = h;
                HA[o] = h;
            }
        }
    }
}

// ---------------------------------------------------------------------------
// APPNP pull iteration: one wave32 per node, lane = output channel.
// acc = dinv[n]^2 * h[n]  (self loop)  + sum_{edges->n} w * h[src]
// h' = 0.9*acc + 0.1*h0
// Node table (12.8 MB) is L2-resident; gathers are coalesced 128B rows.
// ---------------------------------------------------------------------------
__global__ __launch_bounds__(256)
void appnp_step(const float* __restrict__ hin, const float* __restrict__ h0,
                const int* __restrict__ rowptr, const int2* __restrict__ edges,
                const float* __restrict__ dinv, float* __restrict__ hout, int N) {
    int gid  = blockIdx.x * blockDim.x + threadIdx.x;
    int node = gid >> 5;
    int lane = gid & 31;
    if (node >= N) return;

    float di  = dinv[node];
    float acc = di * di * hin[(size_t)node * OUT_CH + lane];

    int beg = rowptr[node];
    int end = rowptr[node + 1];
    for (int j = beg; j < end; ++j) {
        int2 e = edges[j];                       // (src, weight bits) in one b64 load
        acc = fmaf(__int_as_float(e.y), hin[(size_t)e.x * OUT_CH + lane], acc);
    }
    hout[(size_t)node * OUT_CH + lane] =
        fmaf(1.0f - ALPHA, acc, ALPHA * h0[(size_t)node * OUT_CH + lane]);
}

// ---------------------------------------------------------------------------
// Launch
// ---------------------------------------------------------------------------
static inline char* align_up(char* p, size_t a) {
    return (char*)(((uintptr_t)p + a - 1) & ~(uintptr_t)(a - 1));
}

extern "C" void kernel_launch(void* const* d_in, const int* in_sizes, int n_in,
                              void* d_out, int out_size, void* d_ws, size_t ws_size,
                              hipStream_t stream) {
    const float* x  = (const float*)d_in[0];
    const int*   ei = (const int*)d_in[1];
    const float* W1 = (const float*)d_in[2];
    const float* b1 = (const float*)d_in[3];
    const float* W2 = (const float*)d_in[4];
    const float* b2 = (const float*)d_in[5];

    const int N = in_sizes[0] / IN_CH;   // 100000
    const int E = in_sizes[1] / 2;       // 1600000
    const int* rowE = ei;                // edge_index[0]
    const int* colE = ei + E;            // edge_index[1]

    // --- carve workspace ---
    char* p = (char*)d_ws;
    auto take = [&](size_t bytes) -> void* {
        p = align_up(p, 256);
        void* r = (void*)p;
        p += bytes;
        return r;
    };
    f16*   W1T    = (f16*)take((size_t)IN_CH * HID_CH * sizeof(f16));
    f16*   W2T    = (f16*)take((size_t)HID_CH * OUT_CH * sizeof(f16));
    float* deg    = (float*)take((size_t)N * 4);
    float* dinv   = (float*)take((size_t)N * 4);
    int*   cnt    = (int*)take((size_t)N * 4);
    int*   rowptr = (int*)take((size_t)(N + 1) * 4);
    int*   cursor = (int*)take((size_t)N * 4);
    int*   bsum   = (int*)take(4096);
    int*   boff   = (int*)take(4096);
    int2*  edges  = (int2*)take((size_t)E * sizeof(int2));
    float* H0     = (float*)take((size_t)N * OUT_CH * 4);
    float* HA     = (float*)take((size_t)N * OUT_CH * 4);
    float* HB     = (float*)take((size_t)N * OUT_CH * 4);

    const int nbN = (N + 255) / 256;
    const int nbE = (E + 255) / 256;

    // weights -> f16 column-major
    transpose_weights<<<(IN_CH * HID_CH + 255) / 256, 256, 0, stream>>>(W1, W2, W1T, W2T);

    // degrees + norm
    init_deg<<<nbN, 256, 0, stream>>>(deg, cnt, N);
    count_edges<<<nbE, 256, 0, stream>>>(colE, deg, cnt, E);
    compute_dinv<<<nbN, 256, 0, stream>>>(deg, dinv, N);

    // CSR build (bucket by destination)
    scan_blocks<<<nbN, 256, 0, stream>>>(cnt, rowptr, bsum, N);
    scan_sums<<<1, 512, 0, stream>>>(bsum, boff, nbN);
    add_offsets<<<nbN, 256, 0, stream>>>(rowptr, boff, cursor, N, E);
    fill_csr<<<nbE, 256, 0, stream>>>(rowE, colE, dinv, cursor, edges, E);

    // fused MLP: 80 rows per block (100000 = 1250 * 80; clamp handles any N)
    mlp_wmma<<<(N + ROWS - 1) / ROWS, 512, 0, stream>>>(x, W1T, W2T, b1, b2, H0, HA, N);

    // K=10 propagation steps, ping-pong; final step writes d_out
    const int pgrid = ((N * 32) + 255) / 256;
    float* hin = HA;
    for (int t = 0; t < K_ITER; ++t) {
        float* hout = (t == K_ITER - 1) ? (float*)d_out : ((t & 1) ? HA : HB);
        appnp_step<<<pgrid, 256, 0, stream>>>(hin, H0, rowptr, edges, dinv, hout, N);
        hin = hout;
    }
}
